// LSTMBackbone_14207751815618
// MI455X (gfx1250) — matrix operand order repaired and checked by reading, not compiled, not run on hardware
//
#include <hip/hip_runtime.h>
#include <hip/hip_bf16.h>
#include <stdint.h>

// Problem dims (from reference): B=64, T=512, IN_DIM=1024, D=512, L=2
#define BDIM   64
#define TDIM   512
#define INDIM  1024
#define DDIM   512
#define LDIM   2

#define NBLK_PER_LAYER 32        // one block per 16-col n-tile of D
#define NBLK_TOTAL     (2 * NBLK_PER_LAYER)

// CDNA5 async global->LDS path (guarded; falls back to load+ds_store).
#if defined(__has_builtin)
#  if __has_builtin(__builtin_amdgcn_global_load_async_to_lds_b128)
#    define HAVE_ASYNC_LDS 1
#  endif
#  if __has_builtin(__builtin_amdgcn_s_wait_asynccnt)
#    define HAVE_WAIT_ASYNC_BUILTIN 1
#  endif
#endif

typedef __attribute__((ext_vector_type(16))) __bf16 v16bf;
typedef __attribute__((ext_vector_type(8)))  float  v8f;

#if defined(HAVE_ASYNC_LDS)
// The builtin's params are GCC-style int vectors (V4i), AS-qualified:
//   arg0: V4i in global (AS1 / __device__) space, arg1: V4i in LDS (AS3).
typedef int b128v __attribute__((vector_size(16)));
typedef __attribute__((address_space(1))) b128v gb128;
typedef __attribute__((address_space(3))) b128v lb128;
#endif

union FragU { uint4 q[2]; v16bf v; };

// ---------------------------------------------------------------------------
// Fragment loaders for V_WMMA_F32_16X16X32_BF16 (wave32).
// A 16x32 (MxK) bf16 (ISA 7.12.2): lane<16 -> M=lane, K={k0..+7, k0+16..+23}
//                                  lane>=16 -> M=lane-16, K={k0+8..+15, k0+24..+31}
// ---------------------------------------------------------------------------
__device__ __forceinline__ v16bf load_a_bf16(const __bf16* __restrict__ base, int row0,
                                             long rowStride, int k0, int lane) {
  int m    = row0 + (lane & 15);
  int koff = (lane < 16) ? 0 : 8;
  const __bf16* p = base + (long)m * rowStride + k0 + koff;
  FragU u;
  u.q[0] = *(const uint4*)(p);
  u.q[1] = *(const uint4*)(p + 16);
  return u.v;
}

// Same A fragment, fp32 source converted to bf16 in-register (projection only).
__device__ __forceinline__ v16bf load_a_f32(const float* __restrict__ base, int row0,
                                            long rowStride, int k0, int lane) {
  int m    = row0 + (lane & 15);
  int koff = (lane < 16) ? 0 : 8;
  const float* p = base + (long)m * rowStride + k0 + koff;
  float4 a0 = ((const float4*)p)[0];
  float4 a1 = ((const float4*)p)[1];
  float4 b0 = ((const float4*)(p + 16))[0];
  float4 b1 = ((const float4*)(p + 16))[1];
  v16bf v;
  v[0]  = (__bf16)a0.x; v[1]  = (__bf16)a0.y; v[2]  = (__bf16)a0.z; v[3]  = (__bf16)a0.w;
  v[4]  = (__bf16)a1.x; v[5]  = (__bf16)a1.y; v[6]  = (__bf16)a1.z; v[7]  = (__bf16)a1.w;
  v[8]  = (__bf16)b0.x; v[9]  = (__bf16)b0.y; v[10] = (__bf16)b0.z; v[11] = (__bf16)b0.w;
  v[12] = (__bf16)b1.x; v[13] = (__bf16)b1.y; v[14] = (__bf16)b1.z; v[15] = (__bf16)b1.w;
  return v;
}

// B 32x16 (KxN) from a row-major (N,K) weight slice (B = W^T):
// lane<16 -> N=lane, K={k0..+7, k0+16..+23} ; lane>=16 -> N=lane-16, K={k0+8..+15, k0+24..+31}.
__device__ __forceinline__ v16bf load_b_bf16(const __bf16* __restrict__ wbase, int n0,
                                             long rowStride, int k0, int lane) {
  int n    = n0 + (lane & 15);
  int koff = (lane < 16) ? 0 : 8;
  const __bf16* p = wbase + (long)n * rowStride + k0 + koff;
  FragU u;
  u.q[0] = *(const uint4*)(p);
  u.q[1] = *(const uint4*)(p + 16);
  return u.v;
}

// B fragment from an LDS-resident weight slice: rows are [gate*16 + n-local],
// contiguous K of DDIM. Emits ds_load_b128 pairs.
__device__ __forceinline__ v16bf lds_b_frag(const __bf16 (*s)[DDIM], int g, int k0, int lane) {
  int r    = (g << 4) + (lane & 15);
  int koff = (lane < 16) ? 0 : 8;
  const __bf16* p = &s[r][k0 + koff];
  FragU u;
  u.q[0] = *(const uint4*)(p);
  u.q[1] = *(const uint4*)(p + 16);
  return u.v;
}

__device__ __forceinline__ float sigmoidf(float x) {
  return 1.0f / (1.0f + __expf(-x));
}

__device__ __forceinline__ void wait_async_zero() {
#if defined(HAVE_ASYNC_LDS)
#  if defined(HAVE_WAIT_ASYNC_BUILTIN)
  __builtin_amdgcn_s_wait_asynccnt(0);
#  else
  asm volatile("s_wait_asynccnt 0x0" ::: "memory");
#  endif
#endif
}

// ---------------------------------------------------------------------------
// Projection: h_in(BT,D) = x(BT,IN) @ proj_w(D,IN)^T + proj_b   (output bf16)
// Fully parallel GEMM; one wave per 16x16 tile, acc seeded with bias.
// ---------------------------------------------------------------------------
__global__ void proj_kernel(const float*  __restrict__ x,
                            const __bf16* __restrict__ wbf,
                            const float*  __restrict__ bias,
                            __bf16*       __restrict__ hin) {
  int wid  = (blockIdx.x * blockDim.x + threadIdx.x) >> 5;
  int lane = threadIdx.x & 31;
  int nt = wid & 31;          // 32 tiles over N=D
  int mt = wid >> 5;          // 2048 tiles over M=B*T
  int m0 = mt * 16, n0 = nt * 16;
  int nl = n0 + (lane & 15);

  float bv = bias[nl];
  v8f acc;
#pragma unroll
  for (int i = 0; i < 8; ++i) acc[i] = bv;

  for (int k0 = 0; k0 < INDIM; k0 += 32) {
    v16bf a = load_a_f32(x, m0, (long)INDIM, k0, lane);
    v16bf b = load_b_bf16(wbf, n0, (long)INDIM, k0, lane);
    acc = __builtin_amdgcn_wmma_f32_16x16x32_bf16(false, a, false, b,
                                                  (short)0, acc, false, false);
  }
  int mrow = m0 + ((lane >> 4) << 3);
#pragma unroll
  for (int v = 0; v < 8; ++v)
    hin[(long)(mrow + v) * DDIM + nl] = (__bf16)acc[v];
}

// ---------------------------------------------------------------------------
// Grid-wide barrier for the persistent kernel (monotonic counter; 64 blocks
// are all co-resident: 128KB LDS/block -> 2 blocks/WGP -> 32 WGPs).
// ---------------------------------------------------------------------------
__device__ __forceinline__ void grid_sync(int* bar, int iter) {
  __threadfence();
  __syncthreads();
  if (threadIdx.x == 0) {
    __hip_atomic_fetch_add(bar, 1, __ATOMIC_RELEASE, __HIP_MEMORY_SCOPE_AGENT);
    const int target = NBLK_TOTAL * (iter + 1);
    while (__hip_atomic_load(bar, __ATOMIC_ACQUIRE, __HIP_MEMORY_SCOPE_AGENT) < target) {
      __builtin_amdgcn_s_sleep(1);
    }
  }
  __syncthreads();
  __threadfence();
}

// ---------------------------------------------------------------------------
// Persistent 2-layer LSTM. Blocks 0..31 = layer 0, 32..63 = layer 1; layers
// software-pipelined (layer1 lags one step). Each block owns one 16-col
// n-tile of D; its W_ih/W_hh slices (64 rows x 512 K each) are staged ONCE
// into 128KB of LDS (async direct-to-LDS when available) and re-read every
// step via ds_load_b128. Each lane keeps its 8 cell-state values in VGPRs
// for the entire sequence. h is double-buffered in global by step parity.
// ---------------------------------------------------------------------------
__global__ void lstm_persistent_kernel(
    const __bf16* __restrict__ hin,   // (B,T,D): row b stride T*D, +t*D
    const __bf16* __restrict__ Wih,   // (L,4D,D) bf16
    const __bf16* __restrict__ Whh,   // (L,4D,D) bf16
    const float*  __restrict__ bih,   // (L,4D)
    const float*  __restrict__ bhh,   // (L,4D)
    __bf16*       __restrict__ hbuf,  // (2,L,B,D) bf16, zeroed
    float*        __restrict__ out,   // (B,T,D) fp32
    int*          __restrict__ bar)   // grid barrier counter, zeroed
{
  const int layer = (blockIdx.x >= NBLK_PER_LAYER) ? 1 : 0;
  const int nt    = blockIdx.x & (NBLK_PER_LAYER - 1);
  const int lane  = threadIdx.x & 31;
  const int m0    = (threadIdx.x >> 5) * 16;     // 4 waves cover B=64
  const int n0    = nt * 16;
  const int nl    = n0 + (lane & 15);

  const size_t PAR = (size_t)LDIM * BDIM * DDIM; // parity stride in hbuf
  const size_t LHC = (size_t)BDIM * DDIM;        // layer stride in hbuf

  const __bf16* Wih_l = Wih + (size_t)layer * 4 * DDIM * DDIM;
  const __bf16* Whh_l = Whh + (size_t)layer * 4 * DDIM * DDIM;
  const float*  bih_l = bih + layer * 4 * DDIM;
  const float*  bhh_l = bhh + layer * 4 * DDIM;

  // ---- Stage this block's weight slices into LDS (64 rows x 512 K each) ----
  __shared__ __bf16 sWih[64][DDIM];   // 64KB
  __shared__ __bf16 sWhh[64][DDIM];   // 64KB
  {
    const int QPR = DDIM / 8;                       // uint4 chunks per row
    for (int idx = threadIdx.x; idx < 64 * QPR; idx += blockDim.x) {
      int r = idx / QPR, cq = idx % QPR;
      int g = r >> 4, rr = r & 15;
      long grow = (long)(g * DDIM + n0 + rr) * DDIM + (long)cq * 8;
#if defined(HAVE_ASYNC_LDS)
      __builtin_amdgcn_global_load_async_to_lds_b128(
          (gb128*)(Wih_l + grow), (lb128*)&sWih[r][cq * 8], 0, 0);
      __builtin_amdgcn_global_load_async_to_lds_b128(
          (gb128*)(Whh_l + grow), (lb128*)&sWhh[r][cq * 8], 0, 0);
#else
      ((uint4*)&sWih[r][0])[cq] = *(const uint4*)(Wih_l + grow);
      ((uint4*)&sWhh[r][0])[cq] = *(const uint4*)(Whh_l + grow);
#endif
    }
    wait_async_zero();
  }
  __syncthreads();

  // ---- Static per-lane state ----
  float bias_g[4];
#pragma unroll
  for (int g = 0; g < 4; ++g)
    bias_g[g] = bih_l[g * DDIM + nl] + bhh_l[g * DDIM + nl];

  float creg[8];                                   // cell state lives in VGPRs
#pragma unroll
  for (int v = 0; v < 8; ++v) creg[v] = 0.0f;

  const int mrow = m0 + ((lane >> 4) << 3);

  // ---- Pipelined time loop: layer0 does step s, layer1 does step s-1 ----
  for (int s = 0; s < TDIM + 1; ++s) {
    const bool active = (layer == 0) ? (s < TDIM) : (s >= 1);
    if (active) {
      const int t = (layer == 0) ? s : s - 1;
      const size_t pw = (size_t)(t & 1) * PAR;        // write parity
      const size_t pr = (size_t)((t + 1) & 1) * PAR;  // read parity (t-1)

      const __bf16* inp;
      long inp_stride;
      const __bf16* hprev;
      __bf16* hnext;
      if (layer == 0) {
        inp        = hin + (long)t * DDIM;
        inp_stride = (long)TDIM * DDIM;
        hprev      = hbuf + pr;                       // own h(t-1)
        hnext      = hbuf + pw;
        // Prefetch next step's input rows (this lane's two cachelines of
        // h_in[:, t+1, :]) behind the current step's WMMA chain.
        if (t + 1 < TDIM) {
          const __bf16* nx = hin + (long)(t + 1) * DDIM
                           + (long)(m0 + (lane & 15)) * inp_stride;
          __builtin_prefetch(nx, 0, 1);               // global_prefetch_b8
        }
      } else {
        inp        = hbuf + pw;                       // layer0 h(t), parity t&1
        inp_stride = DDIM;
        hprev      = hbuf + pr + LHC;                 // own h(t-1)
        hnext      = hbuf + pw + LHC;
      }

      v8f acc[4];
#pragma unroll
      for (int g = 0; g < 4; ++g) {
#pragma unroll
        for (int i = 0; i < 8; ++i) acc[g][i] = bias_g[g];
      }

      for (int k0 = 0; k0 < DDIM; k0 += 32) {
        v16bf a = load_a_bf16(inp, m0, inp_stride, k0, lane);
#pragma unroll
        for (int g = 0; g < 4; ++g) {
          v16bf b = lds_b_frag(sWih, g, k0, lane);
          acc[g] = __builtin_amdgcn_wmma_f32_16x16x32_bf16(false, a, false, b,
                                                           (short)0, acc[g], false, false);
        }
        v16bf ah = load_a_bf16(hprev, m0, (long)DDIM, k0, lane);
#pragma unroll
        for (int g = 0; g < 4; ++g) {
          v16bf b = lds_b_frag(sWhh, g, k0, lane);
          acc[g] = __builtin_amdgcn_wmma_f32_16x16x32_bf16(false, ah, false, b,
                                                           (short)0, acc[g], false, false);
        }
      }

#pragma unroll
      for (int v = 0; v < 8; ++v) {
        int m = mrow + v;
        float ig = sigmoidf(acc[0][v]);
        float fg = sigmoidf(acc[1][v]);
        float gg = tanhf(acc[2][v]);
        float og = sigmoidf(acc[3][v]);
        float c  = fg * creg[v] + ig * gg;
        float h  = og * tanhf(c);
        creg[v] = c;
        hnext[(long)m * DDIM + nl] = (__bf16)h;
        if (layer == 1)
          out[((long)m * TDIM + t) * DDIM + nl] = h;
      }
    }
    grid_sync(bar, s);   // all blocks, every iteration
  }
}

// ---------------------------------------------------------------------------
// Utility kernels
// ---------------------------------------------------------------------------
__global__ void f32_to_bf16_kernel(const float* __restrict__ in,
                                   __bf16* __restrict__ out, int n) {
  int i = blockIdx.x * blockDim.x + threadIdx.x;
  if (i < n) out[i] = (__bf16)in[i];
}

__global__ void zero_kernel(uint32_t* __restrict__ p, int n_dwords) {
  int i = blockIdx.x * blockDim.x + threadIdx.x;
  if (i < n_dwords) p[i] = 0u;
}

// ---------------------------------------------------------------------------
extern "C" void kernel_launch(void* const* d_in, const int* in_sizes, int n_in,
                              void* d_out, int out_size, void* d_ws, size_t ws_size,
                              hipStream_t stream) {
  (void)in_sizes; (void)n_in; (void)out_size; (void)ws_size;

  const float* x      = (const float*)d_in[0];   // (B,T,IN)
  const float* proj_w = (const float*)d_in[1];   // (D,IN)
  const float* proj_b = (const float*)d_in[2];   // (D)
  const float* W_ih   = (const float*)d_in[3];   // (L,4D,D)
  const float* W_hh   = (const float*)d_in[4];   // (L,4D,D)
  const float* b_ih   = (const float*)d_in[5];   // (L,4D)
  const float* b_hh   = (const float*)d_in[6];   // (L,4D)
  float* out = (float*)d_out;                    // (B,T,D)

  // Workspace carve-out (256B aligned; ~43MB total)
  char* ws = (char*)d_ws;
  size_t off = 0;
  auto carve = [&](size_t bytes) -> char* {
    char* p = ws + off;
    off += (bytes + 255) & ~(size_t)255;
    return p;
  };
  const int  n_projw = DDIM * INDIM;                 // 524288
  const int  n_wih   = LDIM * 4 * DDIM * DDIM;       // 2097152
  const int  n_whh   = n_wih;
  const long n_hin   = (long)BDIM * TDIM * DDIM;     // 16777216
  const int  n_hbuf  = 2 * LDIM * BDIM * DDIM;       // 131072 (parity double buffer)

  __bf16* projw_bf = (__bf16*)carve((size_t)n_projw * 2);
  __bf16* Wih_bf   = (__bf16*)carve((size_t)n_wih * 2);
  __bf16* Whh_bf   = (__bf16*)carve((size_t)n_whh * 2);
  __bf16* hin_bf   = (__bf16*)carve((size_t)n_hin * 2);
  __bf16* hbuf     = (__bf16*)carve((size_t)n_hbuf * 2);   // 262144 B
  int*    bar      = (int*)   carve(256);                  // contiguous after hbuf

  // 1) Convert weights fp32 -> bf16
  f32_to_bf16_kernel<<<(n_projw + 255) / 256, 256, 0, stream>>>(proj_w, projw_bf, n_projw);
  f32_to_bf16_kernel<<<(n_wih   + 255) / 256, 256, 0, stream>>>(W_ih,   Wih_bf,   n_wih);
  f32_to_bf16_kernel<<<(n_whh   + 255) / 256, 256, 0, stream>>>(W_hh,   Whh_bf,   n_whh);

  // 2) Zero h double-buffer + barrier counter (contiguous dwords)
  {
    int n_dw = (n_hbuf * 2 + 256) / 4;   // hbuf bytes + bar region
    zero_kernel<<<(n_dw + 255) / 256, 256, 0, stream>>>((uint32_t*)hbuf, n_dw);
  }

  // 3) Projection GEMM: 65536 tiles, 8 waves per 256-thread block
  {
    int total_waves = (BDIM * TDIM / 16) * (DDIM / 16);   // 65536
    proj_kernel<<<total_waves / 8, 256, 0, stream>>>(x, projw_bf, proj_b, hin_bf);
  }

  // 4) Persistent pipelined recurrence: single launch for all T steps
  lstm_persistent_kernel<<<NBLK_TOTAL, 128, 0, stream>>>(
      hin_bf, Wih_bf, Whh_bf, b_ih, b_hh, hbuf, out, bar);
}
